// DA3CrossFrameCFDistanceLoss_3350074491451
// MI455X (gfx1250) — compile-verified, matrix operand order checked by reading, not run on hardware
//
#include <hip/hip_runtime.h>
#include <math.h>

// Problem constants (from the reference)
#define BB   2
#define VV   8
#define PP   4096
#define DD   1024
#define EE   16384      // 4 extra frames * 4096
#define NREF 256
#define TK   4

typedef __attribute__((ext_vector_type(16))) __bf16 v16bf;
typedef __attribute__((ext_vector_type(8)))  float  v8f;

union Frag16 { uint4 q[2]; v16bf v; };

static __device__ __forceinline__ unsigned short f2bf(float f) {
    unsigned int u = __float_as_uint(f);
    unsigned int lsb = (u >> 16) & 1u;
    u += 0x7fffu + lsb;                  // round-to-nearest-even
    return (unsigned short)(u >> 16);
}

// ---------------------------------------------------------------------------
// Kernel 1: stream extra_t (teacher frames 1,3,5,7), L2-normalize each row,
// down-convert to bf16. One wave per 1024-elem row.
// ---------------------------------------------------------------------------
__global__ void k_extra_norm_bf16(const float* __restrict__ teacher,
                                  unsigned short* __restrict__ ebf16) {
    int wid  = (int)((blockIdx.x * blockDim.x + threadIdx.x) >> 5);
    int lane = threadIdx.x & 31;
    int b = wid >> 14;            // / 16384
    int e = wid & (EE - 1);
    int f = 2 * (e >> 12) + 1;    // EXTRA_FRAMES = 1,3,5,7
    int p = e & (PP - 1);
    const float4* src = (const float4*)(teacher + ((((size_t)b * VV + f) * PP + p) << 10));

    float4 v[8];
    float ss = 0.f;
#pragma unroll
    for (int i = 0; i < 8; ++i) {
        v[i] = src[i * 32 + lane];
        ss += v[i].x * v[i].x + v[i].y * v[i].y + v[i].z * v[i].z + v[i].w * v[i].w;
    }
#pragma unroll
    for (int m = 16; m >= 1; m >>= 1) ss += __shfl_xor(ss, m, 32);
    float inv = 1.0f / fmaxf(sqrtf(ss), 1e-12f);

    ushort4* dst = (ushort4*)(ebf16 + ((size_t)wid << 10));
#pragma unroll
    for (int i = 0; i < 8; ++i) {
        ushort4 o;
        o.x = f2bf(v[i].x * inv); o.y = f2bf(v[i].y * inv);
        o.z = f2bf(v[i].z * inv); o.w = f2bf(v[i].w * inv);
        dst[i * 32 + lane] = o;
    }
}

// ---------------------------------------------------------------------------
// Kernel 2: gather the small slices. One wave per row. 8 categories x 512 rows:
//  cat0: ref_t f32 (+ bf16 copy for the GEMM A-matrix)   teacher[b,0,ref_perm]
//  cat1: ref_s f32                                       student[b,0,ref_perm]
//  cat2..4: shared_t[p] f32, frames {2,4,6}              teacher[b,2p+2,shared_perm]
//  cat5..7: shared_s[p] f32, frames {1,2,3}              student[b,p+1,shared_perm]
// ---------------------------------------------------------------------------
__global__ void k_gather(const float* __restrict__ teacher,
                         const float* __restrict__ student,
                         const int* __restrict__ ref_perm,
                         const int* __restrict__ shared_perm,
                         float* __restrict__ reft, float* __restrict__ refs,
                         float* __restrict__ sht,  float* __restrict__ shs,
                         unsigned short* __restrict__ aref16) {
    int wid  = (int)((blockIdx.x * blockDim.x + threadIdx.x) >> 5);
    int lane = threadIdx.x & 31;
    int cat = wid >> 9;
    int br  = wid & 511;           // b*256 + r
    int b = br >> 8, r = br & 255;

    const float* src; float* dst; bool wbf = false;
    if (cat == 0) {
        int p = ref_perm[r];
        src = teacher + ((((size_t)b * VV + 0) * PP + p) << 10);
        dst = reft + ((size_t)br << 10); wbf = true;
    } else if (cat == 1) {
        int p = ref_perm[r];
        src = student + ((((size_t)b * 4 + 0) * PP + p) << 10);
        dst = refs + ((size_t)br << 10);
    } else if (cat <= 4) {
        int pr = cat - 2, p = shared_perm[r];
        src = teacher + ((((size_t)b * VV + (2 * pr + 2)) * PP + p) << 10);
        dst = sht + (((size_t)pr * 512 + br) << 10);
    } else {
        int pr = cat - 5, p = shared_perm[r];
        src = student + ((((size_t)b * 4 + (pr + 1)) * PP + p) << 10);
        dst = shs + (((size_t)pr * 512 + br) << 10);
    }

    const float4* s4 = (const float4*)src;
    float4* d4 = (float4*)dst;
    ushort4* a4 = (ushort4*)(aref16 + ((size_t)br << 10));
#pragma unroll
    for (int i = 0; i < 8; ++i) {
        float4 t = s4[i * 32 + lane];
        d4[i * 32 + lane] = t;
        if (wbf) {
            ushort4 o;
            o.x = f2bf(t.x); o.y = f2bf(t.y); o.z = f2bf(t.z); o.w = f2bf(t.w);
            a4[i * 32 + lane] = o;
        }
    }
}

// ---------------------------------------------------------------------------
// Kernel 3: sim[b, r, e] = ref_t[b,r,:] . extra_n[b,e,:]  via bf16 WMMA.
// One wave computes a 16x64 output tile (4 accumulator tiles), K-loop of 32
// steps of K=32, double-buffered so next-step loads overlap current WMMAs.
// wid = ((b*256 + eg)*16 + rtile) so consecutive waves share B fragments (L2).
// ---------------------------------------------------------------------------
__global__ void k_gemm_wmma(const unsigned short* __restrict__ aref16,
                            const unsigned short* __restrict__ ebf16,
                            float* __restrict__ sim) {
    int wid  = (int)((blockIdx.x * blockDim.x + threadIdx.x) >> 5);
    int lane = threadIdx.x & 31;
    int rtile = wid & 15;
    int eg    = (wid >> 4) & 255;
    int b     = wid >> 12;

    int mrow = lane & 15;
    int ksel = lane >> 4;                         // 0 or 1 -> K base 0 or 8

    const unsigned short* Arow = aref16 + (((size_t)(b * NREF + rtile * 16 + mrow)) << 10);
    const unsigned short* B0   = ebf16 + ((size_t)b << 24)           // b * EE * DD
                                 + (((size_t)(eg * 64 + mrow)) << 10);

    v8f zero = {0,0,0,0,0,0,0,0};
    v8f acc[4] = {zero, zero, zero, zero};

    // double-buffered fragments: [2] x (A + 4 B)
    Frag16 a[2], bf[2][4];

    {   // preload k-step 0
        int k0 = ksel * 8;
        a[0].q[0] = *(const uint4*)(Arow + k0);
        a[0].q[1] = *(const uint4*)(Arow + k0 + 16);
#pragma unroll
        for (int j = 0; j < 4; ++j) {
            bf[0][j].q[0] = *(const uint4*)(B0 + (size_t)j * 16384 + k0);
            bf[0][j].q[1] = *(const uint4*)(B0 + (size_t)j * 16384 + k0 + 16);
        }
    }

#pragma unroll 2
    for (int kk = 0; kk < 32; ++kk) {
        int cur = kk & 1, nxt = cur ^ 1;
        if (kk < 31) {                       // prefetch k-step kk+1
            int k1 = (kk + 1) * 32 + ksel * 8;
            a[nxt].q[0] = *(const uint4*)(Arow + k1);
            a[nxt].q[1] = *(const uint4*)(Arow + k1 + 16);
#pragma unroll
            for (int j = 0; j < 4; ++j) {
                bf[nxt][j].q[0] = *(const uint4*)(B0 + (size_t)j * 16384 + k1);
                bf[nxt][j].q[1] = *(const uint4*)(B0 + (size_t)j * 16384 + k1 + 16);
            }
        }
#pragma unroll
        for (int j = 0; j < 4; ++j) {
            acc[j] = __builtin_amdgcn_wmma_f32_16x16x32_bf16(
                false, a[cur].v, false, bf[cur][j].v, (short)0, acc[j], false, false);
        }
    }

    // C/D layout: VGPR vg, lane l -> M = vg + (l>=16 ? 8 : 0), N = l&15
    int nc = lane & 15;
    int mo = (lane >> 4) * 8;
#pragma unroll
    for (int j = 0; j < 4; ++j) {
        size_t col = (size_t)(eg * 64 + j * 16 + nc);
#pragma unroll
        for (int vg = 0; vg < 8; ++vg) {
            size_t row = (size_t)(b * NREF + rtile * 16 + mo + vg);
            sim[row * EE + col] = acc[j][vg];
        }
    }
}

// ---------------------------------------------------------------------------
// Kernel 4: per (b,r) row: top-4 over 16384 sim values, then gather the 4
// matching extra_t rows (original fp32 teacher data) into sim_high.
// One 128-thread block per row.
// ---------------------------------------------------------------------------
static __device__ __forceinline__ void top4_insert(float x, int idx, float v[4], int ix[4]) {
    if (x > v[3]) {
        v[3] = x; ix[3] = idx;
#pragma unroll
        for (int k = 3; k > 0; --k) {
            if (v[k] > v[k - 1]) {
                float tv = v[k]; v[k] = v[k - 1]; v[k - 1] = tv;
                int   ti = ix[k]; ix[k] = ix[k - 1]; ix[k - 1] = ti;
            }
        }
    }
}

__global__ void k_topk_gather(const float* __restrict__ sim,
                              const float* __restrict__ teacher,
                              float* __restrict__ simhigh) {
    __shared__ float sv[128 * 4];
    __shared__ int   si[128 * 4];
    __shared__ int   topidx[4];

    int row = blockIdx.x;          // b*256 + r
    int b = row >> 8;
    int t = threadIdx.x;           // 0..127
    const float4* s4 = (const float4*)(sim + (size_t)row * EE);

    float v[4] = {-3.4e38f, -3.4e38f, -3.4e38f, -3.4e38f};
    int   ix[4] = {0, 0, 0, 0};
    for (int i = t; i < EE / 4; i += 128) {     // coalesced b128 scan
        float4 x = s4[i];
        top4_insert(x.x, 4 * i + 0, v, ix);
        top4_insert(x.y, 4 * i + 1, v, ix);
        top4_insert(x.z, 4 * i + 2, v, ix);
        top4_insert(x.w, 4 * i + 3, v, ix);
    }

#pragma unroll
    for (int k = 0; k < 4; ++k) { sv[t * 4 + k] = v[k]; si[t * 4 + k] = ix[k]; }
    __syncthreads();

    for (int stride = 64; stride >= 1; stride >>= 1) {
        if (t < stride) {
            float mv[4]; int mi[4];
#pragma unroll
            for (int k = 0; k < 4; ++k) { mv[k] = sv[t * 4 + k]; mi[k] = si[t * 4 + k]; }
#pragma unroll
            for (int k = 0; k < 4; ++k)
                top4_insert(sv[(t + stride) * 4 + k], si[(t + stride) * 4 + k], mv, mi);
#pragma unroll
            for (int k = 0; k < 4; ++k) { sv[t * 4 + k] = mv[k]; si[t * 4 + k] = mi[k]; }
        }
        __syncthreads();
    }
    if (t < 4) topidx[t] = si[t];
    __syncthreads();

#pragma unroll
    for (int j = 0; j < TK; ++j) {
        int e = topidx[j];
        int f = 2 * (e >> 12) + 1;
        int p = e & (PP - 1);
        const float4* src = (const float4*)(teacher + ((((size_t)b * VV + f) * PP + p) << 10));
        float4* dst = (float4*)(simhigh + (((size_t)row * TK + j) << 10));
        for (int i = t; i < 256; i += 128) dst[i] = src[i];
    }
}

// ---------------------------------------------------------------------------
// Kernel 5: KL + smooth-L1 per row (one wave per 1024-elem row).
// Rows: [0,1536)   d1: p,b,r    (ref - shared)
//       [1536,3584) d2: b,r,j   (ref - sim_high)   [p-independent, counted once]
//       [3584,9728) d3: p,b,r,j (shared - sim_high)
// KL = T/St - (Mt + log St) + (Ms + log Ss),  T = sum e^{xt-Mt}(xt-xs)
// ---------------------------------------------------------------------------
__global__ void k_kl_rows(const float* __restrict__ reft, const float* __restrict__ refs,
                          const float* __restrict__ sht,  const float* __restrict__ shs,
                          const float* __restrict__ simhigh,
                          float* __restrict__ rowloss) {
    int wid  = (int)((blockIdx.x * blockDim.x + threadIdx.x) >> 5);
    int lane = threadIdx.x & 31;

    const float *at, *bt, *as_, *bs;
    if (wid < 1536) {
        int p = wid / 512, br = wid % 512;
        at = reft + ((size_t)br << 10);
        as_ = refs + ((size_t)br << 10);
        bt = sht + (((size_t)p * 512 + br) << 10);
        bs = shs + (((size_t)p * 512 + br) << 10);
    } else if (wid < 3584) {
        int q = wid - 1536;            // b,r,j
        int br = q >> 2, j = q & 3;
        at = reft + ((size_t)br << 10);
        as_ = refs + ((size_t)br << 10);
        bt = bs = simhigh + (((size_t)br * TK + j) << 10);
    } else {
        int q = wid - 3584;
        int p = q / 2048, brj = q % 2048;
        int br = brj >> 2, j = brj & 3;
        at = sht + (((size_t)p * 512 + br) << 10);
        as_ = shs + (((size_t)p * 512 + br) << 10);
        bt = bs = simhigh + (((size_t)br * TK + j) << 10);
    }

    const float4* A4 = (const float4*)at;
    const float4* B4 = (const float4*)bt;
    const float4* C4 = (const float4*)as_;
    const float4* D4 = (const float4*)bs;

    float4 xt[8], xs[8];
    float mt = -3.4e38f, ms = -3.4e38f;
#pragma unroll
    for (int i = 0; i < 8; ++i) {
        float4 a = A4[i * 32 + lane], bb = B4[i * 32 + lane];
        float4 c = C4[i * 32 + lane], dd = D4[i * 32 + lane];
        xt[i].x = a.x - bb.x; xt[i].y = a.y - bb.y; xt[i].z = a.z - bb.z; xt[i].w = a.w - bb.w;
        xs[i].x = c.x - dd.x; xs[i].y = c.y - dd.y; xs[i].z = c.z - dd.z; xs[i].w = c.w - dd.w;
        mt = fmaxf(mt, fmaxf(fmaxf(xt[i].x, xt[i].y), fmaxf(xt[i].z, xt[i].w)));
        ms = fmaxf(ms, fmaxf(fmaxf(xs[i].x, xs[i].y), fmaxf(xs[i].z, xs[i].w)));
    }
#pragma unroll
    for (int m = 16; m >= 1; m >>= 1) {
        mt = fmaxf(mt, __shfl_xor(mt, m, 32));
        ms = fmaxf(ms, __shfl_xor(ms, m, 32));
    }

    float st = 0.f, ss = 0.f, tt = 0.f;
#pragma unroll
    for (int i = 0; i < 8; ++i) {
        float et;
        et = __expf(xt[i].x - mt); st += et; tt += et * (xt[i].x - xs[i].x); ss += __expf(xs[i].x - ms);
        et = __expf(xt[i].y - mt); st += et; tt += et * (xt[i].y - xs[i].y); ss += __expf(xs[i].y - ms);
        et = __expf(xt[i].z - mt); st += et; tt += et * (xt[i].z - xs[i].z); ss += __expf(xs[i].z - ms);
        et = __expf(xt[i].w - mt); st += et; tt += et * (xt[i].w - xs[i].w); ss += __expf(xs[i].w - ms);
    }
#pragma unroll
    for (int m = 16; m >= 1; m >>= 1) {
        st += __shfl_xor(st, m, 32);
        ss += __shfl_xor(ss, m, 32);
        tt += __shfl_xor(tt, m, 32);
    }

    if (lane == 0) {
        float kl = tt / st - (mt + __logf(st)) + (ms + __logf(ss));
        float ax = fabsf(kl);
        float h = (ax < 0.5f) ? (kl * kl) : (ax - 0.25f);   // smooth_l1, beta=0.5
        rowloss[wid] = h;
    }
}

// ---------------------------------------------------------------------------
// Kernel 6: deterministic final reduction (fixed strided order + LDS tree).
// ---------------------------------------------------------------------------
__global__ void k_reduce(const float* __restrict__ rowloss, float* __restrict__ out) {
    __shared__ float acc[256];
    int t = threadIdx.x;
    float s = 0.f;
    for (int i = t; i < 9728; i += 256) {
        float w = (i < 1536) ? (1.0f / 1536.0f)
                 : (i < 3584) ? (1.0f / 2048.0f)
                              : (1.0f / 6144.0f);
        s += rowloss[i] * w;
    }
    acc[t] = s;
    __syncthreads();
    for (int k = 128; k >= 1; k >>= 1) {
        if (t < k) acc[t] += acc[t + k];
        __syncthreads();
    }
    if (t == 0) out[0] = acc[0];
}

// ---------------------------------------------------------------------------
extern "C" void kernel_launch(void* const* d_in, const int* in_sizes, int n_in,
                              void* d_out, int out_size, void* d_ws, size_t ws_size,
                              hipStream_t stream) {
    const float* teacher     = (const float*)d_in[0];
    const float* student     = (const float*)d_in[1];
    const int*   ref_perm    = (const int*)d_in[2];
    const int*   shared_perm = (const int*)d_in[3];
    float* out = (float*)d_out;

    char* w = (char*)d_ws;
    size_t o = 0;
    unsigned short* ebf16  = (unsigned short*)(w + o); o += (size_t)BB * EE * DD * 2;        // 64 MB
    unsigned short* aref16 = (unsigned short*)(w + o); o += (size_t)BB * NREF * DD * 2;      // 1 MB
    float* sim     = (float*)(w + o); o += (size_t)BB * NREF * EE * 4;                       // 32 MB
    float* reft    = (float*)(w + o); o += (size_t)BB * NREF * DD * 4;                       // 2 MB
    float* refs    = (float*)(w + o); o += (size_t)BB * NREF * DD * 4;                       // 2 MB
    float* sht     = (float*)(w + o); o += (size_t)3 * BB * NREF * DD * 4;                   // 6 MB
    float* shs     = (float*)(w + o); o += (size_t)3 * BB * NREF * DD * 4;                   // 6 MB
    float* simhigh = (float*)(w + o); o += (size_t)BB * NREF * TK * DD * 4;                  // 8 MB
    float* rowloss = (float*)(w + o); o += (size_t)9728 * 4;

    // 1. normalize + bf16-convert extra_t: 32768 rows, 1 wave each
    k_extra_norm_bf16<<<4096, 256, 0, stream>>>(teacher, ebf16);
    // 2. gather small slices: 4096 rows
    k_gather<<<512, 256, 0, stream>>>(teacher, student, ref_perm, shared_perm,
                                      reft, refs, sht, shs, aref16);
    // 3. bf16 WMMA GEMM: 8192 waves, each 16x64 output, double-buffered
    k_gemm_wmma<<<1024, 256, 0, stream>>>(aref16, ebf16, sim);
    // 4. top-4 + gather sim_high: one block per (b,r)
    k_topk_gather<<<512, 128, 0, stream>>>(sim, teacher, simhigh);
    // 5. KL + huber: 9728 rows, 1 wave each
    k_kl_rows<<<1216, 256, 0, stream>>>(reft, refs, sht, shs, simhigh, rowloss);
    // 6. deterministic weighted sum -> scalar
    k_reduce<<<1, 256, 0, stream>>>(rowloss, out);
}
